// GCN_71734543778059
// MI455X (gfx1250) — matrix operand order repaired and checked by reading, not compiled, not run on hardware
//
#include <hip/hip_runtime.h>
#include <hip/hip_bf16.h>

#define NODES 100000
#define EDGES 1600000
#define F_IN 256
#define F_HID 128
#define F_OUT 40
#define F_OUT_PAD 48

typedef __bf16 bf16_t;
typedef __bf16 v16bf __attribute__((ext_vector_type(16)));
typedef __bf16 v8bf  __attribute__((ext_vector_type(8)));
typedef __bf16 v4bf  __attribute__((ext_vector_type(4)));
typedef float  v8f   __attribute__((ext_vector_type(8)));

// ---------------- small utility kernels ----------------

__global__ void k_fill_f32(float* __restrict__ p, float v, int n) {
  int i = blockIdx.x * blockDim.x + threadIdx.x;
  if (i < n) p[i] = v;
}

__global__ void k_count_edges(const int* __restrict__ dst, float* __restrict__ deg, int e) {
  int i = blockIdx.x * blockDim.x + threadIdx.x;
  if (i < e) atomicAdd(&deg[dst[i]], 1.0f);
}

__global__ void k_rsqrt_inplace(float* __restrict__ p, int n) {
  int i = blockIdx.x * blockDim.x + threadIdx.x;
  if (i < n) p[i] = rsqrtf(p[i]);
}

__global__ void k_f32_to_bf16_v4(const float4* __restrict__ in, v4bf* __restrict__ out, int n4) {
  int i = blockIdx.x * blockDim.x + threadIdx.x;
  if (i >= n4) return;
  float4 v = in[i];
  v4bf o = { (__bf16)v.x, (__bf16)v.y, (__bf16)v.z, (__bf16)v.w };
  out[i] = o;
}

__global__ void k_relu_to_bf16_v4(const float4* __restrict__ in, v4bf* __restrict__ out, int n4) {
  int i = blockIdx.x * blockDim.x + threadIdx.x;
  if (i >= n4) return;
  float4 v = in[i];
  v4bf o = { (__bf16)fmaxf(v.x, 0.0f), (__bf16)fmaxf(v.y, 0.0f),
             (__bf16)fmaxf(v.z, 0.0f), (__bf16)fmaxf(v.w, 0.0f) };
  out[i] = o;
}

// Pack row-major W[K x N] (f32) into the CDNA5 WMMA B-fragment layout:
// PW[((kt*tilesN + tn)*32 + lane)*16 + j] = bf16(W[(kt*32 + (lane>>4)*8 + (j&7) + (j>>3)*16)*N + tn*16 + (lane&15)])
// so each lane's 16-element B column fragment is one contiguous 32B chunk.
__global__ void k_pack_w(const float* __restrict__ W, bf16_t* __restrict__ PW,
                         int K, int N, int tilesN) {
  int idx = blockIdx.x * blockDim.x + threadIdx.x;
  int total = K * tilesN * 16;
  if (idx >= total) return;
  int j  = idx & 15;
  int l  = (idx >> 4) & 31;
  int t  = idx >> 9;           // kt*tilesN + tn
  int tn = t % tilesN;
  int kt = t / tilesN;
  int k  = kt * 32 + ((l >> 4) * 8) + (j & 7) + ((j >> 3) * 16);
  int n  = tn * 16 + (l & 15);
  float v = (n < N) ? W[(size_t)k * N + n] : 0.0f;
  PW[idx] = (bf16_t)v;
}

// ---------------- fused WMMA GEMM + self-loop/bias epilogue ----------------
// One wave computes a 16 x (TN*16) row strip: the A fragment is loaded once
// per 32-deep k-step and reused for TN WMMAs (8x arithmetic intensity vs a
// single 16x16 tile).  Epilogue writes raw H into G (edge-gather source) and
// AGG = H * dinv^2 + bias (aggregation accumulator seed) in one pass.
template<int K, int TN, int F, int LDG>
__global__ void k_gemm_fused(const bf16_t* __restrict__ A,
                             const bf16_t* __restrict__ PW,
                             const float* __restrict__ dinv,
                             const float* __restrict__ bias,
                             float* __restrict__ G,
                             float* __restrict__ AGG,
                             int tilesM) {
  const int lane = threadIdx.x & 31;
  int tm = blockIdx.x * (blockDim.x >> 5) + (threadIdx.x >> 5);
  if (tm >= tilesM) return;                  // uniform per wave: EXEC stays all-ones

  // A fragment: lane holds row (lane&15); K slots {b..b+7, b+16..b+23}, b=(lane>>4)*8
  const bf16_t* ap = A + (size_t)(tm * 16 + (lane & 15)) * K + ((lane >> 4) * 8);

  v8f acc[TN];
#pragma unroll
  for (int tn = 0; tn < TN; ++tn)
    acc[tn] = (v8f){0.f, 0.f, 0.f, 0.f, 0.f, 0.f, 0.f, 0.f};

#pragma unroll
  for (int kt = 0; kt < (K >> 5); ++kt) {
    v8bf alo = *(const v8bf*)(ap);           // K = base .. base+7   (b128 load)
    v8bf ahi = *(const v8bf*)(ap + 16);      // K = base+16..base+23 (b128 load)
    v16bf a = __builtin_shufflevector(alo, ahi,
        0, 1, 2, 3, 4, 5, 6, 7, 8, 9, 10, 11, 12, 13, 14, 15);
    // B fragments: lane holds column (lane&15) of the 32x16 tile, K slots as for A, 16 contiguous bf16
    const bf16_t* bp = PW + ((size_t)(kt * TN) * 32 + lane) * 16;
#pragma unroll
    for (int tn = 0; tn < TN; ++tn) {
      v16bf b = *(const v16bf*)(bp + (size_t)tn * 512);
      acc[tn] = __builtin_amdgcn_wmma_f32_16x16x32_bf16(
          /*neg_a=*/false, a, /*neg_b=*/false, b,
          /*c_mod=*/(short)0, acc[tn], /*reuse_a=*/false, /*reuse_b=*/false);
    }
    ap += 32;
  }

  // D layout: VGPR i -> row tm*16 + i + (lane>=16 ? 8 : 0), col tn*16 + (lane&15)
  const int colbase = lane & 15;
  const int rbase = tm * 16 + ((lane >> 4) << 3);
  float bv[TN];
#pragma unroll
  for (int tn = 0; tn < TN; ++tn) {
    int c = tn * 16 + colbase;
    bv[tn] = (c < F) ? bias[c] : 0.0f;
  }
#pragma unroll
  for (int i = 0; i < 8; ++i) {
    int r = rbase + i;
    float di = dinv[r];
    float di2 = di * di;
#pragma unroll
    for (int tn = 0; tn < TN; ++tn) {
      int c = tn * 16 + colbase;
      float h = acc[tn][i];
      G[(size_t)r * LDG + c] = h;
      if (c < F) AGG[(size_t)r * F + c] = h * di2 + bv[tn];
    }
  }
}

// ---------------- edge scatter ---------------------------------------------
// Grid-stride wave32 per edge: float4 gather of H[src,:], f32 atomic scatter
// to AGG[dst,:], with a one-iteration-ahead prefetch of the next source row.
__global__ void k_agg_edges(const float* __restrict__ H, const float* __restrict__ dinv,
                            const int* __restrict__ src, const int* __restrict__ dst,
                            float* __restrict__ out, int F, int ldh, int E) {
  const int lane = threadIdx.x & 31;
  const int wavesPerBlock = blockDim.x >> 5;
  const int wavesTotal = gridDim.x * wavesPerBlock;
  int w = blockIdx.x * wavesPerBlock + (threadIdx.x >> 5);
  const int nf4 = F >> 2;
  for (int e = w; e < E; e += wavesTotal) {
    int en = e + wavesTotal;
    if (en < E) {
      int sn = src[en];
      __builtin_prefetch(H + (size_t)sn * ldh + (size_t)lane * 4, 0, 3);
    }
    int s = src[e];
    int d = dst[e];
    float coef = dinv[s] * dinv[d];
    const float4* h4 = (const float4*)(H + (size_t)s * ldh);
    float* orow = out + (size_t)d * F;
    for (int f4 = lane; f4 < nf4; f4 += 32) {
      float4 v = h4[f4];
      atomicAdd(&orow[f4 * 4 + 0], v.x * coef);
      atomicAdd(&orow[f4 * 4 + 1], v.y * coef);
      atomicAdd(&orow[f4 * 4 + 2], v.z * coef);
      atomicAdd(&orow[f4 * 4 + 3], v.w * coef);
    }
  }
}

__global__ void k_log_softmax(const float* __restrict__ in, float* __restrict__ out,
                              int n, int C) {
  int i = blockIdx.x * blockDim.x + threadIdx.x;
  if (i >= n) return;
  const float* r = in + (size_t)i * C;
  float m = r[0];
  for (int c = 1; c < C; ++c) m = fmaxf(m, r[c]);
  float s = 0.0f;
  for (int c = 0; c < C; ++c) s += expf(r[c] - m);
  float ls = logf(s);
  float* o = out + (size_t)i * C;
  for (int c = 0; c < C; ++c) o[c] = r[c] - m - ls;
}

// ---------------- host orchestration --------------------------------------

extern "C" void kernel_launch(void* const* d_in, const int* in_sizes, int n_in,
                              void* d_out, int out_size, void* d_ws, size_t ws_size,
                              hipStream_t stream) {
  (void)in_sizes; (void)n_in; (void)out_size; (void)ws_size;

  const float* x  = (const float*)d_in[0];
  const int*   ei = (const int*)d_in[1];     // [2, E]: row0 = src, row1 = dst
  const float* W1 = (const float*)d_in[2];
  const float* b1 = (const float*)d_in[3];
  const float* W2 = (const float*)d_in[4];
  const float* b2 = (const float*)d_in[5];
  const float* W3 = (const float*)d_in[6];
  const float* b3 = (const float*)d_in[7];
  const int* src = ei;
  const int* dst = ei + EDGES;

  char* ws = (char*)d_ws;
  auto carve = [&](size_t bytes) -> char* {
    char* p = ws;
    ws += (bytes + 255) & ~(size_t)255;
    return p;
  };
  float*  dinv = (float*) carve((size_t)NODES * 4);
  bf16_t* Abf  = (bf16_t*)carve((size_t)NODES * F_IN * 2);      // bf16 GEMM input
  float*  G    = (float*) carve((size_t)NODES * F_HID * 4);     // raw H (gather src)
  float*  AGG  = (float*) carve((size_t)NODES * F_HID * 4);     // aggregation accum
  bf16_t* PW1  = (bf16_t*)carve((size_t)F_IN  * F_HID * 2);
  bf16_t* PW2  = (bf16_t*)carve((size_t)F_HID * F_HID * 2);
  bf16_t* PW3  = (bf16_t*)carve((size_t)F_HID * F_OUT_PAD * 2);

  const int T = 256;
  const int tilesM = NODES / 16;             // 6250 exactly
  const int gemmBlocks = (tilesM + 3) / 4;   // 4 waves / block
  const int edgeBlocks = 50000;              // 400000 waves, 4 edges each

  // degrees: deg = incoming-edge count + 1 (self loop), dinv = rsqrt(deg)
  k_fill_f32<<<(NODES + T - 1) / T, T, 0, stream>>>(dinv, 1.0f, NODES);
  k_count_edges<<<(EDGES + T - 1) / T, T, 0, stream>>>(dst, dinv, EDGES);
  k_rsqrt_inplace<<<(NODES + T - 1) / T, T, 0, stream>>>(dinv, NODES);

  // pack weights into WMMA B-fragment layout (bf16)
  k_pack_w<<<(F_IN  * 8 * 16 + T - 1) / T, T, 0, stream>>>(W1, PW1, F_IN,  F_HID, 8);
  k_pack_w<<<(F_HID * 8 * 16 + T - 1) / T, T, 0, stream>>>(W2, PW2, F_HID, F_HID, 8);
  k_pack_w<<<(F_HID * 3 * 16 + T - 1) / T, T, 0, stream>>>(W3, PW3, F_HID, F_OUT, 3);

  // x -> bf16
  {
    int n4 = NODES * F_IN / 4;
    k_f32_to_bf16_v4<<<(n4 + T - 1) / T, T, 0, stream>>>((const float4*)x, (v4bf*)Abf, n4);
  }

  // ---- layer 1: h = relu(norm-agg(x @ W1) + b1) ----
  {
    k_gemm_fused<F_IN, 8, F_HID, F_HID><<<gemmBlocks, 128, 0, stream>>>(
        Abf, PW1, dinv, b1, G, AGG, tilesM);
    k_agg_edges<<<edgeBlocks, 256, 0, stream>>>(G, dinv, src, dst, AGG, F_HID, F_HID, EDGES);
    int n4 = NODES * F_HID / 4;
    k_relu_to_bf16_v4<<<(n4 + T - 1) / T, T, 0, stream>>>((const float4*)AGG, (v4bf*)Abf, n4);
  }

  // ---- layer 2 ----
  {
    k_gemm_fused<F_HID, 8, F_HID, F_HID><<<gemmBlocks, 128, 0, stream>>>(
        Abf, PW2, dinv, b2, G, AGG, tilesM);
    k_agg_edges<<<edgeBlocks, 256, 0, stream>>>(G, dinv, src, dst, AGG, F_HID, F_HID, EDGES);
    int n4 = NODES * F_HID / 4;
    k_relu_to_bf16_v4<<<(n4 + T - 1) / T, T, 0, stream>>>((const float4*)AGG, (v4bf*)Abf, n4);
  }

  // ---- layer 3 (N padded 40 -> 48 for 16-wide tiles) + log_softmax ----
  {
    k_gemm_fused<F_HID, 3, F_OUT, F_OUT_PAD><<<gemmBlocks, 128, 0, stream>>>(
        Abf, PW3, dinv, b3, G, AGG, tilesM);
    k_agg_edges<<<edgeBlocks, 256, 0, stream>>>(G, dinv, src, dst, AGG, F_OUT, F_OUT_PAD, EDGES);
    k_log_softmax<<<(NODES + T - 1) / T, T, 0, stream>>>(AGG, (float*)d_out, NODES, F_OUT);
  }
}
